// TimeMixingResBlock_78795470012781
// MI455X (gfx1250) — matrix operand order repaired and checked by reading, not compiled, not run on hardware
//
#include <hip/hip_runtime.h>
#include <hip/hip_bf16.h>

typedef __bf16 bf16;
typedef __attribute__((ext_vector_type(8)))  bf16  v8bf;
typedef __attribute__((ext_vector_type(16))) bf16  v16bf;
typedef __attribute__((ext_vector_type(8)))  float v8f;

#define B_   64
#define T_   512
#define F_   1024
#define NTOT (64 * 512 * 1024)
#define RED_BLOCKS 1024

// ---------------- pass 1: per-block partial sums (deterministic, no atomics) ----
__global__ __launch_bounds__(256) void reduce_partial(const float* __restrict__ x,
                                                      float2* __restrict__ part) {
    __shared__ float2 red[256];
    const float4* x4 = (const float4*)x;
    const int n4 = NTOT / 4;
    float s = 0.f, ss = 0.f;
    for (int i = blockIdx.x * 256 + threadIdx.x; i < n4; i += RED_BLOCKS * 256) {
        float4 v = x4[i];
        s  += v.x + v.y + v.z + v.w;
        ss += v.x * v.x + v.y * v.y + v.z * v.z + v.w * v.w;
    }
    red[threadIdx.x] = make_float2(s, ss);
    __syncthreads();
    for (int off = 128; off > 0; off >>= 1) {
        if (threadIdx.x < off) {
            red[threadIdx.x].x += red[threadIdx.x + off].x;
            red[threadIdx.x].y += red[threadIdx.x + off].y;
        }
        __syncthreads();
    }
    if (threadIdx.x == 0) part[blockIdx.x] = red[0];
}

// ---------------- pass 2: finalize scale/shift -------------------------------
__global__ __launch_bounds__(256) void reduce_final(const float2* __restrict__ part,
                                                    const float* __restrict__ gamma,
                                                    const float* __restrict__ beta,
                                                    float* __restrict__ sc) {
    __shared__ float2 red[256];
    float s = 0.f, ss = 0.f;
    for (int i = threadIdx.x; i < RED_BLOCKS; i += 256) { s += part[i].x; ss += part[i].y; }
    red[threadIdx.x] = make_float2(s, ss);
    __syncthreads();
    for (int off = 128; off > 0; off >>= 1) {
        if (threadIdx.x < off) {
            red[threadIdx.x].x += red[threadIdx.x + off].x;
            red[threadIdx.x].y += red[threadIdx.x + off].y;
        }
        __syncthreads();
    }
    if (threadIdx.x == 0) {
        float inv   = 1.0f / (float)NTOT;
        float mean  = red[0].x * inv;
        float var   = red[0].y * inv - mean * mean;
        float rsig  = rsqrtf(var + 1e-5f);
        float scale = rsig * gamma[0];
        sc[0] = scale;
        sc[1] = beta[0] - mean * scale;   // y = x*scale + shift
    }
}

// ---------------- W fp32 -> bf16 (512KB, L2-resident afterwards) -------------
__global__ __launch_bounds__(256) void wconv(const float* __restrict__ W,
                                             bf16* __restrict__ W16) {
    int i = blockIdx.x * 256 + threadIdx.x;
    W16[i] = (bf16)W[i];
}

// ---------------- fused normalize + batched GEMM (WMMA) + bias/relu/residual -
// out[b,u,f] = x[b,u,f] + relu(bias[u] + sum_t W[u,t] * (x[b,t,f]*scale+shift))
// Block: 256 thr = 8 waves, tile 128(u) x 64(f); wave = 32x32; K step = 32.
// Double-buffered LDS; A tile staged with async global->LDS DMA (ASYNCcnt).
__global__ __launch_bounds__(256) void gemm_fused(const float* __restrict__ x,
                                                  const bf16* __restrict__ W16,
                                                  const float* __restrict__ bias,
                                                  const float* __restrict__ sc,
                                                  float* __restrict__ out) {
    // strides padded to 48 halfs (96B = 24 banks: rotates, 16B aligned)
    __shared__ __align__(32) bf16 Alds[2][128 * 48];   // W[u_local][k_local]
    __shared__ __align__(32) bf16 Blds[2][64 * 48];    // y transposed [f_local][k_local]

    const int b  = blockIdx.z;
    const int u0 = blockIdx.y * 128;
    const int f0 = blockIdx.x * 64;

    const float scale = sc[0];
    const float shift = sc[1];

    const int lane = threadIdx.x & 31;
    const int wave = threadIdx.x >> 5;
    const int msub = (wave & 3) * 32;     // wave's u offset within tile
    const int nsub = (wave >> 2) * 32;    // wave's f offset within tile
    const int lrow = lane & 15;
    const int hi   = lane >> 4;           // lane half-group

    // staging coordinates
    const int arow = threadIdx.x >> 1;            // 0..127
    const int acol = (threadIdx.x & 1) << 4;      // 0 or 16
    const int brow = threadIdx.x >> 3;            // t-local 0..31
    const int bcol = (threadIdx.x & 7) << 3;      // f-local 0,8,..,56

    v8f acc[2][2] = {};

    // ---- stage helpers (inlined by macro-ish lambdas) ----
    auto stageA_async = [&](int p, int kk) {
        const bf16* g = W16 + ((size_t)(u0 + arow) << 9) + kk + acol;
        unsigned int l = (unsigned int)(size_t)(&Alds[p][arow * 48 + acol]);
        unsigned long long ga = (unsigned long long)(size_t)g;
        // DMA 2x16B global->LDS; offset: applies to BOTH lds and mem address.
        asm volatile("global_load_async_to_lds_b128 %0, %1, off"
                     :: "v"(l), "v"(ga) : "memory");
        asm volatile("global_load_async_to_lds_b128 %0, %1, off offset:16"
                     :: "v"(l), "v"(ga) : "memory");
    };
    auto stageB = [&](int p, int kk) {
        const float* g = x + ((size_t)b * T_ + (kk + brow)) * F_ + f0 + bcol;
        float4 p0 = *(const float4*)g;
        float4 p1 = *(const float4*)(g + 4);
        float v[8] = {p0.x, p0.y, p0.z, p0.w, p1.x, p1.y, p1.z, p1.w};
#pragma unroll
        for (int j = 0; j < 8; ++j)
            Blds[p][(bcol + j) * 48 + brow] = (bf16)fmaf(v[j], scale, shift);
    };

    // ---- prologue: fill buffer 0 ----
    stageA_async(0, 0);
    stageB(0, 0);

    for (int kk = 0; kk < 512; kk += 32) {
        const int p = (kk >> 5) & 1;
        // our async A-DMA for tile p must have landed in LDS
        asm volatile("s_wait_asynccnt 0x0" ::: "memory");
        // all waves: staging complete (and all reads of buffer p^1 done)
        __syncthreads();

        if (kk + 32 < 512) {               // prefetch next tile into other buffer
            stageA_async(p ^ 1, kk + 32);
            stageB(p ^ 1, kk + 32);
        }

        // --- fragments per documented CDNA5 16-bit layouts -------------------
        // A 16x32: lanes0-15 elems0-7 -> K0..7, elems8-15 -> K16..23; lanes16-31: +8
        v16bf afr[2], bfr[2];
#pragma unroll
        for (int mi = 0; mi < 2; ++mi) {
            const bf16* pa = &Alds[p][(msub + mi * 16 + lrow) * 48 + hi * 8];
            v8bf lo = *(const v8bf*)pa;
            v8bf hh = *(const v8bf*)(pa + 16);
            afr[mi] = __builtin_shufflevector(lo, hh, 0,1,2,3,4,5,6,7,8,9,10,11,12,13,14,15);
        }
        // B 32x16: N=lane%16; lanes0-15 elems0-7 -> K0..7, elems8-15 -> K16..23; lanes16-31: +8
#pragma unroll
        for (int ni = 0; ni < 2; ++ni) {
            const bf16* pb = &Blds[p][(nsub + ni * 16 + lrow) * 48 + hi * 8];
            v8bf blo = *(const v8bf*)pb;
            v8bf bhh = *(const v8bf*)(pb + 16);
            bfr[ni] = __builtin_shufflevector(blo, bhh, 0,1,2,3,4,5,6,7,8,9,10,11,12,13,14,15);
        }
#pragma unroll
        for (int mi = 0; mi < 2; ++mi)
#pragma unroll
            for (int ni = 0; ni < 2; ++ni)
                acc[mi][ni] = __builtin_amdgcn_wmma_f32_16x16x32_bf16(
                    false, afr[mi], false, bfr[ni], (short)0, acc[mi][ni], false, false);
    }

    // --- epilogue: bias + relu + residual. D layout: VGPR r, lanes0-15 M=r,
    // lanes16-31 M=r+8, N=lane%16 -> per-r stores are 16 consecutive floats.
#pragma unroll
    for (int mi = 0; mi < 2; ++mi)
#pragma unroll
        for (int ni = 0; ni < 2; ++ni)
#pragma unroll
            for (int r = 0; r < 8; ++r) {
                int urow = u0 + msub + mi * 16 + r + hi * 8;
                int fcol = f0 + nsub + ni * 16 + lrow;
                size_t idx = ((size_t)b * T_ + urow) * F_ + fcol;
                float h = acc[mi][ni][r] + bias[urow];
                out[idx] = x[idx] + fmaxf(h, 0.0f);
            }
}

extern "C" void kernel_launch(void* const* d_in, const int* in_sizes, int n_in,
                              void* d_out, int out_size, void* d_ws, size_t ws_size,
                              hipStream_t stream) {
    const float* x     = (const float*)d_in[0];
    const float* W     = (const float*)d_in[1];
    const float* bias  = (const float*)d_in[2];
    const float* gamma = (const float*)d_in[3];
    const float* beta  = (const float*)d_in[4];
    float* out = (float*)d_out;

    char* ws = (char*)d_ws;
    float2* part = (float2*)ws;                  // [0, 8192)
    float*  sc   = (float*)(ws + 8192);          // scale, shift
    bf16*   W16  = (bf16*)(ws + 8704);           // 512KB bf16 weights

    reduce_partial<<<RED_BLOCKS, 256, 0, stream>>>(x, part);
    reduce_final<<<1, 256, 0, stream>>>(part, gamma, beta, sc);
    wconv<<<(512 * 512) / 256, 256, 0, stream>>>(W, W16);

    dim3 grid(F_ / 64, T_ / 128, B_);            // (16, 4, 64) = 4096 blocks
    gemm_fused<<<grid, 256, 0, stream>>>(x, W16, bias, sc, out);
}
